// LinearFlexQMixer_20770461843876
// MI455X (gfx1250) — matrix operand order repaired and checked by reading, not compiled, not run on hardware
//
#include <hip/hip_runtime.h>
#include <math.h>

#define NA   16
#define NE_  128
#define ED_  96
#define HE_  128
#define MD_  32
#define NH_  4
#define HD_  32

typedef __bf16 bf16_t;
typedef __attribute__((ext_vector_type(16))) __bf16 v16bf;
typedef __attribute__((ext_vector_type(8)))  __bf16 v8bf;
typedef __attribute__((ext_vector_type(4)))  __bf16 v4bf;
typedef __attribute__((ext_vector_type(8)))  float  v8f;

#define WMMA_BF16(a, b, c) \
  __builtin_amdgcn_wmma_f32_16x16x32_bf16(false, (a), false, (b), (short)0, (c), false, false)

#define CAT16(lo, hi) \
  __builtin_shufflevector((lo), (hi), 0, 1, 2, 3, 4, 5, 6, 7, 8, 9, 10, 11, 12, 13, 14, 15)

// ---- CDNA5 WMMA fragment loaders (wave32 layouts per ISA 7.12.2) ----
// All loads are 2x ds_load_b128 (sources arranged so each lane reads contiguous K).

// A matrix 16x32 bf16, row-major [m][k] in LDS.
// lane 0-15: M=lane, K = [kb,kb+8)+[kb+16,kb+24), kb=0; lanes 16-31: kb=8.
__device__ __forceinline__ v16bf lds_load_a(const bf16_t* A, int ld, int row0, int k0, int lane) {
  int m  = lane & 15;
  int kb = (lane >> 4) << 3;
  const bf16_t* r = A + (row0 + m) * ld + k0 + kb;
  v8bf lo = *(const v8bf*)(r);
  v8bf hi = *(const v8bf*)(r + 16);
  return CAT16(lo, hi);
}

// B matrix 32x16 (KxN) where the LDS source is laid out [n][k] contiguous in k
// (transposed weights, or row-major K-matrix for q@k^T).
// lane 0-15: N=lane, K = [kb,kb+8)+[kb+16,kb+24), kb=0; lanes 16-31: kb=8.
__device__ __forceinline__ v16bf lds_load_bk(const bf16_t* Wt, int ldK, int k0, int col0, int lane) {
  int n  = lane & 15;
  int kb = (lane >> 4) << 3;
  const bf16_t* r = Wt + (col0 + n) * ldK + k0 + kb;
  v8bf lo = *(const v8bf*)(r);
  v8bf hi = *(const v8bf*)(r + 16);
  return CAT16(lo, hi);
}

// LDS byte offsets (163728 B dynamic LDS -> 2 workgroups per 320 KB WGP)
#define SM_ENT   0        // bf16 128x96   (24576)  -> reused: sWsm(16K) + sOut(4K)
#define SM_WGT   24576    // bf16 <=128x128 (32768) transposed weight staging [n][k]
#define SM_X1    57344    // bf16 128x128  (32768)  -> reused as f32 logits 4x16x128
#define SM_K     90112    // bf16 128x128  row-major [key][d]
#define SM_V     122880   // bf16 128x128  transposed [d][key]
#define SM_Q     155648   // bf16 16x128   (4096)   -> reused as attn
#define SM_EF    159744   // f32  128
#define SM_RED   160256   // f32  64
#define SM_X3    160512   // f32  16x32    (2048)
#define SM_SC    162560   // f32  scalars
#define SM_BIAS  162576   // f32  288 (fc1_b 128 | out_b 128 | fc2_b 32)
#define SM_TOTAL 163728

__global__ void __launch_bounds__(256) qmix_wmma_kernel(
    const float* __restrict__ qs, const float* __restrict__ ents,
    const unsigned char* __restrict__ emask,
    const float* __restrict__ w1_fc1_w, const float* __restrict__ w1_fc1_b,
    const float* __restrict__ w1_in_w,  const float* __restrict__ w1_out_w,
    const float* __restrict__ w1_out_b, const float* __restrict__ w1_fc2_w,
    const float* __restrict__ w1_fc2_b,
    const float* __restrict__ v_fc1_w, const float* __restrict__ v_fc1_b,
    const float* __restrict__ v_in_w,  const float* __restrict__ v_out_w,
    const float* __restrict__ v_out_b, const float* __restrict__ v_fc2_w,
    const float* __restrict__ v_fc2_b,
    float* __restrict__ out)
{
  extern __shared__ unsigned char smem[];
  bf16_t* sEnt  = (bf16_t*)(smem + SM_ENT);
  bf16_t* sWgtT = (bf16_t*)(smem + SM_WGT);
  bf16_t* sX1   = (bf16_t*)(smem + SM_X1);
  float*  sLog  = (float*) (smem + SM_X1);   // reuse after x1 consumed
  bf16_t* sK    = (bf16_t*)(smem + SM_K);
  bf16_t* sVT   = (bf16_t*)(smem + SM_V);
  bf16_t* sQ    = (bf16_t*)(smem + SM_Q);    // reused as attn
  float*  sEF   = (float*) (smem + SM_EF);
  float*  sRed  = (float*) (smem + SM_RED);
  float*  sX3   = (float*) (smem + SM_X3);
  float*  sSc   = (float*) (smem + SM_SC);
  float*  sBias = (float*) (smem + SM_BIAS);
  bf16_t* sWsm  = (bf16_t*)(smem + SM_ENT);          // 4*16*128 bf16 = 16384 B
  bf16_t* sOut  = (bf16_t*)(smem + SM_ENT + 16384);  // 16*128 bf16  = 4096 B

  const int tid  = threadIdx.x;
  const int lane = tid & 31;
  const int wid  = tid >> 5;
  const int b    = blockIdx.x;

  const float* entsRow = ents + (size_t)b * (NE_ * ED_);
  const unsigned char* maskRow = emask + (size_t)b * NE_;

  // ---- preamble: entity mask -> LDS once; all(ef) via wave32 ballot ----
  if (tid < NE_) sEF[tid] = maskRow[tid] ? 1.0f : 0.0f;
  __syncthreads();
  if (tid < 32) {
    bool ok = true;
#pragma unroll
    for (int t = 0; t < 4; ++t) ok = ok && (sEF[tid + 32 * t] > 0.5f);
    unsigned long long bal = __ballot(ok);
    if (tid == 0) sSc[0] = ((bal & 0xFFFFFFFFull) == 0xFFFFFFFFull) ? 1.0f : 0.0f;
  }

  float qtot = 0.0f;

  for (int p = 0; p < 2; ++p) {
    const float* fc1_w = p ? v_fc1_w : w1_fc1_w;
    const float* fc1_b = p ? v_fc1_b : w1_fc1_b;
    const float* in_w  = p ? v_in_w  : w1_in_w;
    const float* out_w = p ? v_out_w : w1_out_w;
    const float* out_b = p ? v_out_b : w1_out_b;
    const float* fc2_w = p ? v_fc2_w : w1_fc2_w;
    const float* fc2_b = p ? v_fc2_b : w1_fc2_b;

    // ---- P0: stage entities (float4 loads -> packed bf16x4 stores),
    //          fc1_w transposed [n][k], and the three biases ----
    for (int idx = tid; idx < (NE_ * ED_) / 4; idx += 256) {
      float4 f = ((const float4*)entsRow)[idx];
      v4bf pk = {(bf16_t)f.x, (bf16_t)f.y, (bf16_t)f.z, (bf16_t)f.w};
      *(v4bf*)(sEnt + idx * 4) = pk;
    }
    for (int idx = tid; idx < HE_ * (ED_ / 4); idx += 256) {
      int n = idx & (HE_ - 1);
      int k = (idx >> 7) << 2;
      v4bf pk = {(bf16_t)fc1_w[(k + 0) * HE_ + n], (bf16_t)fc1_w[(k + 1) * HE_ + n],
                 (bf16_t)fc1_w[(k + 2) * HE_ + n], (bf16_t)fc1_w[(k + 3) * HE_ + n]};
      *(v4bf*)(sWgtT + n * ED_ + k) = pk;
    }
    if (tid < 128) sBias[tid] = fc1_b[tid];
    else           sBias[tid] = out_b[tid - 128];
    if (tid < 32)  sBias[256 + tid] = fc2_b[tid];
    __builtin_prefetch(in_w + (size_t)tid * 192, 0, 0);  // warm qkv weights in L2
    __syncthreads();

    // ---- P1: x1 = relu(ents @ fc1_w + b)  (128x96 @ 96x128) ----
    {
      v16bf a0 = lds_load_a(sEnt, ED_, 16 * wid, 0,  lane);
      v16bf a1 = lds_load_a(sEnt, ED_, 16 * wid, 32, lane);
      v16bf a2 = lds_load_a(sEnt, ED_, 16 * wid, 64, lane);
      int n = lane & 15, mhi = (lane >> 4) << 3;
      for (int nt = 0; nt < 8; ++nt) {
        v8f acc = {};
        acc = WMMA_BF16(a0, lds_load_bk(sWgtT, ED_, 0,  16 * nt, lane), acc);
        acc = WMMA_BF16(a1, lds_load_bk(sWgtT, ED_, 32, 16 * nt, lane), acc);
        acc = WMMA_BF16(a2, lds_load_bk(sWgtT, ED_, 64, 16 * nt, lane), acc);
        float bias = sBias[16 * nt + n];
#pragma unroll
        for (int r = 0; r < 8; ++r) {
          float x = acc[r] + bias;
          x = x > 0.0f ? x : 0.0f;
          sX1[(16 * wid + mhi + r) * HE_ + 16 * nt + n] = (bf16_t)x;
        }
      }
    }
    __syncthreads();

    // ---- P2: qkv = x1 @ in_w, 3 chunks of 128 cols; q pre-scaled by 1/sqrt(HD).
    //          K chunk stored row-major [key][d]; V chunk stored transposed [d][key]. ----
    const float qscale = 0.17677669529663687f;
    for (int c = 0; c < 3; ++c) {
      for (int idx = tid; idx < HE_ * (HE_ / 4); idx += 256) {
        int n = idx & (HE_ - 1);
        int k = (idx >> 7) << 2;
        v4bf pk = {(bf16_t)in_w[(k + 0) * 384 + c * 128 + n],
                   (bf16_t)in_w[(k + 1) * 384 + c * 128 + n],
                   (bf16_t)in_w[(k + 2) * 384 + c * 128 + n],
                   (bf16_t)in_w[(k + 3) * 384 + c * 128 + n]};
        *(v4bf*)(sWgtT + n * HE_ + k) = pk;
      }
      __syncthreads();
      int n = lane & 15, mhi = (lane >> 4) << 3;
      if (c == 0) {  // only 16 q rows needed: 8 N-tiles, one per wave
        int nt = wid;
        v8f acc = {};
#pragma unroll
        for (int ks = 0; ks < 4; ++ks) {
          v16bf a = lds_load_a(sX1, HE_, 0, 32 * ks, lane);
          acc = WMMA_BF16(a, lds_load_bk(sWgtT, HE_, 32 * ks, 16 * nt, lane), acc);
        }
#pragma unroll
        for (int r = 0; r < 8; ++r)
          sQ[(mhi + r) * HE_ + 16 * nt + n] = (bf16_t)(acc[r] * qscale);
      } else {
        v16bf a0 = lds_load_a(sX1, HE_, 16 * wid, 0,  lane);
        v16bf a1 = lds_load_a(sX1, HE_, 16 * wid, 32, lane);
        v16bf a2 = lds_load_a(sX1, HE_, 16 * wid, 64, lane);
        v16bf a3 = lds_load_a(sX1, HE_, 16 * wid, 96, lane);
        for (int nt = 0; nt < 8; ++nt) {
          v8f acc = {};
          acc = WMMA_BF16(a0, lds_load_bk(sWgtT, HE_, 0,  16 * nt, lane), acc);
          acc = WMMA_BF16(a1, lds_load_bk(sWgtT, HE_, 32, 16 * nt, lane), acc);
          acc = WMMA_BF16(a2, lds_load_bk(sWgtT, HE_, 64, 16 * nt, lane), acc);
          acc = WMMA_BF16(a3, lds_load_bk(sWgtT, HE_, 96, 16 * nt, lane), acc);
          if (c == 1) {  // K: row-major [key][d]
#pragma unroll
            for (int r = 0; r < 8; ++r)
              sK[(16 * wid + mhi + r) * HE_ + 16 * nt + n] = (bf16_t)acc[r];
          } else {       // V: transposed [d][key]; rows contiguous -> 2x b64 stores
            v4bf lo = {(bf16_t)acc[0], (bf16_t)acc[1], (bf16_t)acc[2], (bf16_t)acc[3]};
            v4bf hi = {(bf16_t)acc[4], (bf16_t)acc[5], (bf16_t)acc[6], (bf16_t)acc[7]};
            bf16_t* d0 = sVT + (16 * nt + n) * HE_ + 16 * wid + mhi;
            *(v4bf*)(d0)     = lo;
            *(v4bf*)(d0 + 4) = hi;
          }
        }
      }
      __syncthreads();
    }

    // ---- P3: logits = q @ k^T per head (masked) -> f32 LDS ----
    {
      int n = lane & 15, mhi = (lane >> 4) << 3;
#pragma unroll
      for (int i = 0; i < 4; ++i) {
        int t = wid * 4 + i;
        int h = t >> 3, nt = t & 7;
        v16bf a  = lds_load_a (sQ, HE_, 0, 32 * h, lane);
        v16bf bt = lds_load_bk(sK, HE_, 32 * h, 16 * nt, lane);  // rows of sK = contiguous d
        v8f acc = {};
        acc = WMMA_BF16(a, bt, acc);
#pragma unroll
        for (int r = 0; r < 8; ++r) {
          int q = mhi + r, key = 16 * nt + n;
          bool pre = (sEF[q] > 0.5f) || (sEF[key] > 0.5f);
          sLog[(h * 16 + q) * NE_ + key] = pre ? -1.0e9f : acc[r];
        }
      }
    }
    __builtin_prefetch(out_w + (size_t)tid * 64, 0, 0);
    __syncthreads();

    // ---- P4: softmax over 128 keys; 4 lanes per row, 64 rows, all 256 threads.
    //          Vector LDS reads, exp cached in regs, shfl_xor cross-lane reduce. ----
    {
      int row = tid >> 2;   // (h*16+q)
      int sub = tid & 3;
      int q = row & 15;
      const float4* rp = (const float4*)(sLog + row * NE_ + sub * 32);
      float vbuf[32];
#pragma unroll
      for (int i = 0; i < 8; ++i) {
        float4 f = rp[i];
        vbuf[4 * i + 0] = f.x; vbuf[4 * i + 1] = f.y;
        vbuf[4 * i + 2] = f.z; vbuf[4 * i + 3] = f.w;
      }
      float mx = -3.0e38f;
#pragma unroll
      for (int i = 0; i < 32; ++i) mx = fmaxf(mx, vbuf[i]);
      mx = fmaxf(mx, __shfl_xor(mx, 1, 32));
      mx = fmaxf(mx, __shfl_xor(mx, 2, 32));
      float s = 0.0f;
#pragma unroll
      for (int i = 0; i < 32; ++i) { vbuf[i] = __expf(vbuf[i] - mx); s += vbuf[i]; }
      s += __shfl_xor(s, 1, 32);
      s += __shfl_xor(s, 2, 32);
      bool allm = (sEF[q] > 0.5f) || (sSc[0] > 0.5f);
      float inv = allm ? 0.0f : 1.0f / s;
      bf16_t* wp = sWsm + row * NE_ + sub * 32;
#pragma unroll
      for (int i = 0; i < 8; ++i) {
        v4bf pk = {(bf16_t)(vbuf[4 * i + 0] * inv), (bf16_t)(vbuf[4 * i + 1] * inv),
                   (bf16_t)(vbuf[4 * i + 2] * inv), (bf16_t)(vbuf[4 * i + 3] * inv)};
        *(v4bf*)(wp + 4 * i) = pk;
      }
    }
    __syncthreads();

    // ---- P5: attn = w @ v per head -> sAttn 16x128 (one 16-col tile per wave) ----
    {
      int h = wid >> 1, half = wid & 1;
      int n = lane & 15, mhi = (lane >> 4) << 3;
      const bf16_t* A = sWsm + (h * 16) * NE_;
      v8f acc = {};
#pragma unroll
      for (int ks = 0; ks < 4; ++ks) {
        v16bf a = lds_load_a(A, NE_, 0, 32 * ks, lane);
        acc = WMMA_BF16(a, lds_load_bk(sVT, HE_, 32 * ks, 32 * h + 16 * half, lane), acc);
      }
      bf16_t* sAttn = sQ;  // reuse
#pragma unroll
      for (int r = 0; r < 8; ++r)
        sAttn[(mhi + r) * HE_ + 32 * h + 16 * half + n] = (bf16_t)acc[r];
    }
    __syncthreads();

    // ---- P6: out = attn @ out_w + out_b, masked rows -> 0 ----
    for (int idx = tid; idx < HE_ * (HE_ / 4); idx += 256) {
      int n = idx & (HE_ - 1);
      int k = (idx >> 7) << 2;
      v4bf pk = {(bf16_t)out_w[(k + 0) * HE_ + n], (bf16_t)out_w[(k + 1) * HE_ + n],
                 (bf16_t)out_w[(k + 2) * HE_ + n], (bf16_t)out_w[(k + 3) * HE_ + n]};
      *(v4bf*)(sWgtT + n * HE_ + k) = pk;
    }
    __syncthreads();
    {
      int nt = wid;
      int n = lane & 15, mhi = (lane >> 4) << 3;
      const bf16_t* sAttn = sQ;
      v8f acc = {};
#pragma unroll
      for (int ks = 0; ks < 4; ++ks) {
        v16bf a = lds_load_a(sAttn, HE_, 0, 32 * ks, lane);
        acc = WMMA_BF16(a, lds_load_bk(sWgtT, HE_, 32 * ks, 16 * nt, lane), acc);
      }
      float bias = sBias[128 + 16 * nt + n];
#pragma unroll
      for (int r = 0; r < 8; ++r) {
        int q = mhi + r;
        float x = (sEF[q] > 0.5f) ? 0.0f : (acc[r] + bias);
        sOut[q * HE_ + 16 * nt + n] = (bf16_t)x;
      }
    }
    __syncthreads();

    // ---- P7: x3 = out @ fc2_w + fc2_b, masked rows -> 0  (16x128 @ 128x32) ----
    for (int idx = tid; idx < MD_ * (HE_ / 4); idx += 256) {
      int n = idx & (MD_ - 1);
      int k = (idx >> 5) << 2;
      v4bf pk = {(bf16_t)fc2_w[(k + 0) * MD_ + n], (bf16_t)fc2_w[(k + 1) * MD_ + n],
                 (bf16_t)fc2_w[(k + 2) * MD_ + n], (bf16_t)fc2_w[(k + 3) * MD_ + n]};
      *(v4bf*)(sWgtT + n * HE_ + k) = pk;
    }
    __syncthreads();
    if (wid < 2) {
      int nt = wid;
      int n = lane & 15, mhi = (lane >> 4) << 3;
      v8f acc = {};
#pragma unroll
      for (int ks = 0; ks < 4; ++ks) {
        v16bf a = lds_load_a(sOut, HE_, 0, 32 * ks, lane);
        acc = WMMA_BF16(a, lds_load_bk(sWgtT, HE_, 32 * ks, 16 * nt, lane), acc);
      }
      float bias = sBias[256 + 16 * nt + n];
#pragma unroll
      for (int r = 0; r < 8; ++r) {
        int q = mhi + r;
        float x = (sEF[q] > 0.5f) ? 0.0f : (acc[r] + bias);
        sX3[q * MD_ + 16 * nt + n] = x;
      }
    }
    __syncthreads();

    // ---- P8: per-agent sums (4 lanes per agent), then mix ----
    if (tid < 64) {
      int a = tid >> 2, sub = tid & 3;
      const float* xr = sX3 + a * MD_ + sub * 8;
      float s = 0.0f;
#pragma unroll
      for (int i = 0; i < 8; ++i) s += xr[i];
      s += __shfl_xor(s, 1, 32);
      s += __shfl_xor(s, 2, 32);
      if (sub == 0) sRed[a] = s;
    }
    __syncthreads();
    if (tid == 0) {
      if (p == 0) {
        for (int a = 0; a < NA; ++a)
          qtot += qs[(size_t)b * NA + a] * fabsf(sRed[a] * (1.0f / MD_));
      } else {
        float s = 0.0f;
        for (int a = 0; a < NA; ++a) s += sRed[a];
        qtot += s * (1.0f / (NA * MD_));
      }
    }
    __syncthreads();
  }

  if (tid == 0) out[b] = qtot;
}

extern "C" void kernel_launch(void* const* d_in, const int* in_sizes, int n_in,
                              void* d_out, int out_size, void* d_ws, size_t ws_size,
                              hipStream_t stream) {
  const float* qs   = (const float*)d_in[0];
  const float* ents = (const float*)d_in[1];
  const unsigned char* em = (const unsigned char*)d_in[2];
  const float* w1_fc1_w = (const float*)d_in[3];
  const float* w1_fc1_b = (const float*)d_in[4];
  const float* w1_in_w  = (const float*)d_in[5];
  const float* w1_out_w = (const float*)d_in[6];
  const float* w1_out_b = (const float*)d_in[7];
  const float* w1_fc2_w = (const float*)d_in[8];
  const float* w1_fc2_b = (const float*)d_in[9];
  const float* v_fc1_w  = (const float*)d_in[10];
  const float* v_fc1_b  = (const float*)d_in[11];
  const float* v_in_w   = (const float*)d_in[12];
  const float* v_out_w  = (const float*)d_in[13];
  const float* v_out_b  = (const float*)d_in[14];
  const float* v_fc2_w  = (const float*)d_in[15];
  const float* v_fc2_b  = (const float*)d_in[16];
  float* out = (float*)d_out;

  int B = in_sizes[1] / (NE_ * ED_);  // 1600

  (void)hipFuncSetAttribute((const void*)qmix_wmma_kernel,
                            hipFuncAttributeMaxDynamicSharedMemorySize, SM_TOTAL);

  qmix_wmma_kernel<<<dim3(B), dim3(256), SM_TOTAL, stream>>>(
      qs, ents, em,
      w1_fc1_w, w1_fc1_b, w1_in_w, w1_out_w, w1_out_b, w1_fc2_w, w1_fc2_b,
      v_fc1_w, v_fc1_b, v_in_w, v_out_w, v_out_b, v_fc2_w, v_fc2_b,
      out);
}
